// LabelSelfAttnMeanPooling_89945205113618
// MI455X (gfx1250) — matrix operand order repaired and checked by reading, not compiled, not run on hardware
//
#include <hip/hip_runtime.h>
#include <math.h>

// CDNA5 WMMA types
typedef __attribute__((ext_vector_type(16))) _Float16 v16h;
typedef __attribute__((ext_vector_type(8)))  float    v8f;

// b128 payload type for async global->LDS copies (matches builtin prototype:
// 'int __attribute__((vector_size(4*sizeof(int)))) addrspace(1/3) *')
typedef int vs4i __attribute__((vector_size(16)));
typedef vs4i __attribute__((address_space(1))) vs4i_g;   // global
typedef vs4i __attribute__((address_space(3))) vs4i_l;   // LDS

#define BDIM 4
#define TDIM 2048
#define DDIM 512
#define HDIM 8
#define DH   64
#define BT   (BDIM * TDIM)
#define D3   (3 * DDIM)

// Async global->LDS copy path (CDNA5 GLOBAL_LOAD_ASYNC_TO_LDS_B128, ASYNCcnt)
#if __has_builtin(__builtin_amdgcn_global_load_async_to_lds_b128) && \
    __has_builtin(__builtin_amdgcn_s_wait_asynccnt)
#define USE_ASYNC_LDS 1
#else
#define USE_ASYNC_LDS 0
#endif

// ---- WMMA fragment index maps (ISA 7.12.2, 16-bit operands, wave32) ----
// A-matrix 16x32: lane 0-15 -> M=lane, halves: V0..3 = K 0..7, V4..7 = K 16..23
//                 lane 16-31 -> same M, K offset +8
__device__ __forceinline__ int ka_map(int lane, int h) {
    return (h & 7) | (((lane >> 4) & 1) << 3) | ((h >> 3) << 4);
}
// B-matrix 32x16: same K order as A (lanes 0-15: K 0..7,16..23; lanes 16-31: +8), N=lane
__device__ __forceinline__ int kb_map(int lane, int h) {
    return (h & 7) | (((lane >> 4) & 1) << 3) | ((h >> 3) << 4);
}
// C/D f32 16x16: value in reg r of lane L is (M = r + 8*(L>>4), N = L&15)

__device__ __forceinline__ v8f wmma_f16(v16h a, v16h b, v8f c) {
    return __builtin_amdgcn_wmma_f32_16x16x32_f16(false, a, false, b, (short)0, c,
                                                  false, false);
}

__device__ __forceinline__ v8f zero_v8f() {
    v8f z;
#pragma unroll
    for (int i = 0; i < 8; ++i) z[i] = 0.0f;
    return z;
}

// =====================================================================
// Kernel 1: LayerNorm + zero-token mask + fused score bias. 1 wave/token.
// =====================================================================
__global__ void ln_mask_kernel(const float* __restrict__ embs,
                               const float* __restrict__ labels,
                               const float* __restrict__ ln_g,
                               const float* __restrict__ ln_b,
                               const float* __restrict__ alpha_p,
                               const float* __restrict__ beta_p,
                               _Float16* __restrict__ xh,
                               float* __restrict__ biasArr,
                               float* __restrict__ mArr) {
    int wave  = threadIdx.x >> 5;
    int lane  = threadIdx.x & 31;
    int token = blockIdx.x * 8 + wave;

    const float* src = embs + (size_t)token * DDIM;
    float x[16];
    float s = 0.0f, s2 = 0.0f;
#pragma unroll
    for (int j = 0; j < 4; ++j) {
        float4 v = ((const float4*)src)[lane + 32 * j];
        x[4 * j + 0] = v.x; x[4 * j + 1] = v.y;
        x[4 * j + 2] = v.z; x[4 * j + 3] = v.w;
        s  += v.x + v.y + v.z + v.w;
        s2 += v.x * v.x + v.y * v.y + v.z * v.z + v.w * v.w;
    }
#pragma unroll
    for (int off = 16; off; off >>= 1) {
        s  += __shfl_xor(s, off, 32);
        s2 += __shfl_xor(s2, off, 32);
    }
    float mean = s * (1.0f / DDIM);
    float var  = s2 * (1.0f / DDIM) - mean * mean;
    float rstd = rsqrtf(var + 1e-5f);

    _Float16* drow = xh + (size_t)token * DDIM;
    float asum = 0.0f;
#pragma unroll
    for (int j = 0; j < 4; ++j) {
        union { _Float16 h[4]; uint2 u; } pk;
#pragma unroll
        for (int c = 0; c < 4; ++c) {
            int   d = 4 * (lane + 32 * j) + c;
            float y = (x[4 * j + c] - mean) * rstd * ln_g[d] + ln_b[d];
            asum += fabsf(y);
            pk.h[c] = (_Float16)y;
        }
        ((uint2*)drow)[lane + 32 * j] = pk.u;
    }
#pragma unroll
    for (int off = 16; off; off >>= 1) asum += __shfl_xor(asum, off, 32);

    if (lane == 0) {
        bool kpm = (asum <= 1e-6f);
        biasArr[token] = kpm ? -1e30f
                             : (alpha_p[0] * labels[token] + beta_p[0]);
        mArr[token] = kpm ? 0.0f : 1.0f;
    }
}

// =====================================================================
// Kernel 2: convert in_proj_w f32 -> f16 (one-time per launch)
// =====================================================================
__global__ void cvt_f32_f16_kernel(const float* __restrict__ s,
                                   _Float16* __restrict__ d, int n) {
    int i = blockIdx.x * 256 + threadIdx.x;
    if (i < n) d[i] = (_Float16)s[i];
}

// =====================================================================
// Kernel 3: QKV projection via WMMA. Each wave: 16(M) x 64(N) strip.
// Output scattered to head-major [B*H, T, dh] f16 q/k/v buffers.
// =====================================================================
__global__ void qkv_wmma_kernel(const _Float16* __restrict__ xh,
                                const _Float16* __restrict__ wh,
                                const float* __restrict__ inb,
                                _Float16* __restrict__ qh,
                                _Float16* __restrict__ kh,
                                _Float16* __restrict__ vh) {
    int wave = threadIdx.x >> 5, lane = threadIdx.x & 31;
    int lr = lane & 15, lh = (lane >> 4) & 1;
    int rowbase = blockIdx.x * 128 + wave * 16;
    int colbase = blockIdx.y * 64;

    v8f acc[4];
#pragma unroll
    for (int t = 0; t < 4; ++t) acc[t] = zero_v8f();

    for (int kb = 0; kb < DDIM; kb += 32) {
        v16h a;
#pragma unroll
        for (int h = 0; h < 16; ++h)
            a[h] = xh[(size_t)(rowbase + lr) * DDIM + kb + ka_map(lane, h)];
#pragma unroll
        for (int t = 0; t < 4; ++t) {
            int  ncol = colbase + t * 16 + lr;
            v16h bf;
#pragma unroll
            for (int h = 0; h < 16; ++h)
                bf[h] = wh[(size_t)ncol * DDIM + kb + kb_map(lane, h)];
            acc[t] = wmma_f16(a, bf, acc[t]);
        }
    }

#pragma unroll
    for (int t = 0; t < 4; ++t) {
        int   col   = colbase + t * 16 + lr;
        int   which = col >> 9;          // 0:q 1:k 2:v
        int   d     = col & 511;
        int   head  = d >> 6, dd = d & 63;
        float bias  = inb[col];
        _Float16* dst = (which == 0) ? qh : (which == 1) ? kh : vh;
#pragma unroll
        for (int r = 0; r < 8; ++r) {
            int token = rowbase + r + 8 * lh;
            int b = token >> 11, tt = token & 2047;
            dst[(((size_t)(b * HDIM + head)) * TDIM + tt) * DH + dd] =
                (_Float16)(acc[t][r] + bias);
        }
    }
}

// =====================================================================
// Kernel 4: flash attention + masked pooling partials.
// Block = 8 waves = 128 queries of one (b,h). LDS-staged K/V (128 keys),
// staged via CDNA5 async global->LDS copies when available.
// =====================================================================
__global__ void attn_wmma_kernel(const _Float16* __restrict__ qh,
                                 const _Float16* __restrict__ kh,
                                 const _Float16* __restrict__ vh,
                                 const float* __restrict__ biasArr,
                                 const float* __restrict__ mArr,
                                 float* __restrict__ part) {
    __shared__ __align__(16) _Float16 Klds[128 * DH];
    __shared__ __align__(16) _Float16 Vlds[128 * DH];
    __shared__ __align__(16) _Float16 Plds[8][16 * 32];
    __shared__ float pool_w[8][64];

    int wave = threadIdx.x >> 5, lane = threadIdx.x & 31;
    int lr = lane & 15, lh = (lane >> 4) & 1;
    int bh = blockIdx.x >> 4;                 // (b*H + head)
    int qblk = blockIdx.x & 15;
    int b = bh >> 3, head = bh & 7;
    int qbase = qblk * 128 + wave * 16;
    (void)head;

    const _Float16* Qp = qh + ((size_t)bh * TDIM) * DH;
    const _Float16* Kp = kh + ((size_t)bh * TDIM) * DH;
    const _Float16* Vp = vh + ((size_t)bh * TDIM) * DH;

    // Q fragments (two K=32 chunks over dh=64), pre-scaled by 1/sqrt(dh)
    v16h qa[2];
#pragma unroll
    for (int c = 0; c < 2; ++c)
#pragma unroll
        for (int h = 0; h < 16; ++h)
            qa[c][h] = Qp[(size_t)(qbase + lr) * DH + c * 32 + ka_map(lane, h)] *
                       (_Float16)0.125f;

    float rmax[8], rsum[8];
#pragma unroll
    for (int r = 0; r < 8; ++r) { rmax[r] = -INFINITY; rsum[r] = 0.0f; }
    v8f oacc[4];
#pragma unroll
    for (int t = 0; t < 4; ++t) oacc[t] = zero_v8f();

    for (int kb = 0; kb < TDIM; kb += 128) {
        // ---- stage 128x64 K and V tiles into LDS (contiguous spans) ----
#if USE_ASYNC_LDS
        {
            const _Float16* sk = Kp + (size_t)kb * DH;
            const _Float16* sv = Vp + (size_t)kb * DH;
#pragma unroll
            for (int j = 0; j < 4; ++j) {
                int i = (threadIdx.x + 256 * j) * 8;   // 8 halves = 16 bytes
                __builtin_amdgcn_global_load_async_to_lds_b128(
                    (vs4i_g*)(sk + i), (vs4i_l*)(Klds + i), 0, 0);
                __builtin_amdgcn_global_load_async_to_lds_b128(
                    (vs4i_g*)(sv + i), (vs4i_l*)(Vlds + i), 0, 0);
            }
            __builtin_amdgcn_s_wait_asynccnt(0);
        }
#else
        {
            const float4* sk = (const float4*)(Kp + (size_t)kb * DH);
            const float4* sv = (const float4*)(Vp + (size_t)kb * DH);
            float4* dk = (float4*)Klds;
            float4* dv = (float4*)Vlds;
#pragma unroll
            for (int j = 0; j < 4; ++j) {
                int i = threadIdx.x + 256 * j;
                dk[i] = sk[i];
                dv[i] = sv[i];
            }
        }
#endif
        __syncthreads();

        for (int sub = 0; sub < 4; ++sub) {     // 32 keys per step
            int k0 = sub * 32;
            v8f s0 = zero_v8f(), s1 = zero_v8f();
#pragma unroll
            for (int c = 0; c < 2; ++c) {
                v16h bk0, bk1;
#pragma unroll
                for (int h = 0; h < 16; ++h) {
                    int dco = c * 32 + kb_map(lane, h);
                    bk0[h] = Klds[(k0 + lr) * DH + dco];
                    bk1[h] = Klds[(k0 + 16 + lr) * DH + dco];
                }
                s0 = wmma_f16(qa[c], bk0, s0);
                s1 = wmma_f16(qa[c], bk1, s1);
            }
            // additive label bias / key-padding mask (per key column)
            float bv0 = biasArr[b * TDIM + kb + k0 + lr];
            float bv1 = biasArr[b * TDIM + kb + k0 + 16 + lr];
#pragma unroll
            for (int r = 0; r < 8; ++r) { s0[r] += bv0; s1[r] += bv1; }

            // online softmax (rows live across 16-lane halves)
            float p0[8], p1[8], scale[8];
#pragma unroll
            for (int r = 0; r < 8; ++r) {
                float vm = fmaxf(s0[r], s1[r]);
#pragma unroll
                for (int off = 1; off < 16; off <<= 1)
                    vm = fmaxf(vm, __shfl_xor(vm, off, 32));
                float mnew = fmaxf(rmax[r], vm);
                float sc   = __expf(rmax[r] - mnew);
                rmax[r] = mnew;
                p0[r] = __expf(s0[r] - mnew);
                p1[r] = __expf(s1[r] - mnew);
                float ps = p0[r] + p1[r];
#pragma unroll
                for (int off = 1; off < 16; off <<= 1)
                    ps += __shfl_xor(ps, off, 32);
                rsum[r] = rsum[r] * sc + ps;
                scale[r] = sc;
            }
#pragma unroll
            for (int t = 0; t < 4; ++t)
#pragma unroll
                for (int r = 0; r < 8; ++r) oacc[t][r] *= scale[r];

            // redistribute P (D-layout -> A-layout) via per-wave LDS scratch
            _Float16* pw = &Plds[wave][0];
#pragma unroll
            for (int r = 0; r < 8; ++r) {
                pw[(r + 8 * lh) * 32 + lr]      = (_Float16)p0[r];
                pw[(r + 8 * lh) * 32 + 16 + lr] = (_Float16)p1[r];
            }
            v16h pa;
#pragma unroll
            for (int h = 0; h < 16; ++h)
                pa[h] = pw[lr * 32 + ka_map(lane, h)];

            // O += P @ V  (K dim = 32 keys)
#pragma unroll
            for (int t = 0; t < 4; ++t) {
                v16h bv;
#pragma unroll
                for (int h = 0; h < 16; ++h)
                    bv[h] = Vlds[(k0 + kb_map(lane, h)) * DH + t * 16 + lr];
                oacc[t] = wmma_f16(pa, bv, oacc[t]);
            }
        }
        __syncthreads();
    }

    // finalize: weight rows by mask/rowsum, reduce 16 queries -> 64 dh values
    float w[8];
#pragma unroll
    for (int r = 0; r < 8; ++r)
        w[r] = mArr[b * TDIM + qbase + r + 8 * lh] / rsum[r];

#pragma unroll
    for (int t = 0; t < 4; ++t) {
        float p = 0.0f;
#pragma unroll
        for (int r = 0; r < 8; ++r) {
            float v = oacc[t][r] * w[r];
            if (!isfinite(v)) v = 0.0f;   // nan_to_num safety
            p += v;
        }
        p += __shfl_xor(p, 16, 32);       // combine row halves
        if (lh == 0) pool_w[wave][t * 16 + lr] = p;
    }
    __syncthreads();
    if (threadIdx.x < 64) {               // fixed-order (deterministic) sum
        float s = 0.0f;
#pragma unroll
        for (int wv = 0; wv < 8; ++wv) s += pool_w[wv][threadIdx.x];
        part[(size_t)blockIdx.x * 64 + threadIdx.x] = s;
    }
}

// =====================================================================
// Kernel 5: reduce per-qblock partials -> pooled_pre[B, D]
// =====================================================================
__global__ void pool_reduce_kernel(const float* __restrict__ part,
                                   float* __restrict__ pooled) {
    int idx = blockIdx.x * 256 + threadIdx.x;   // 0..2047
    int bh = idx >> 6, dd = idx & 63;
    int b = bh >> 3, head = bh & 7;
    float s = 0.0f;
#pragma unroll
    for (int q = 0; q < 16; ++q) s += part[(size_t)(bh * 16 + q) * 64 + dd];
    pooled[b * DDIM + head * DH + dd] = s;
}

// =====================================================================
// Kernel 6: tiny fp32 out-projection on the pooled vector + bias + /count
// =====================================================================
__global__ void outproj_kernel(const float* __restrict__ pooled,
                               const float* __restrict__ mArr,
                               const float* __restrict__ out_w,
                               const float* __restrict__ out_b,
                               float* __restrict__ out) {
    __shared__ float red[256];
    int b = blockIdx.x >> 1;
    int n = ((blockIdx.x & 1) * 256) + threadIdx.x;

    float c = 0.0f;
    for (int t = threadIdx.x; t < TDIM; t += 256) c += mArr[b * TDIM + t];
    red[threadIdx.x] = c;
    __syncthreads();
    for (int off = 128; off; off >>= 1) {
        if (threadIdx.x < off) red[threadIdx.x] += red[threadIdx.x + off];
        __syncthreads();
    }
    float count = red[0];

    float acc = 0.0f;
    const float* pr = pooled + b * DDIM;
    const float* wr = out_w + (size_t)n * DDIM;
    for (int k = 0; k < DDIM; ++k) acc += pr[k] * wr[k];

    out[b * DDIM + n] = (count > 0.0f) ? (acc / count + out_b[n]) : 0.0f;
}

// =====================================================================
extern "C" void kernel_launch(void* const* d_in, const int* in_sizes, int n_in,
                              void* d_out, int out_size, void* d_ws, size_t ws_size,
                              hipStream_t stream) {
    const float* embs    = (const float*)d_in[0];
    const float* labels  = (const float*)d_in[1];
    const float* ln_g    = (const float*)d_in[2];
    const float* ln_b    = (const float*)d_in[3];
    const float* in_w    = (const float*)d_in[4];
    const float* in_b    = (const float*)d_in[5];
    const float* out_w   = (const float*)d_in[6];
    const float* out_b   = (const float*)d_in[7];
    const float* alpha_p = (const float*)d_in[8];
    const float* beta_p  = (const float*)d_in[9];

    char* ws = (char*)d_ws;
    const size_t MB = 1024 * 1024;
    _Float16* xh      = (_Float16*)(ws);                       // 8 MB
    _Float16* wh      = (_Float16*)(ws + 8 * MB);              // 1.5 MB
    _Float16* qh      = (_Float16*)(ws + 10 * MB);             // 8 MB
    _Float16* kh      = (_Float16*)(ws + 18 * MB);             // 8 MB
    _Float16* vh      = (_Float16*)(ws + 26 * MB);             // 8 MB
    float*    biasArr = (float*)(ws + 34 * MB);                // 32 KB
    float*    mArr    = (float*)(ws + 34 * MB + 32 * 1024);    // 32 KB
    float*    part    = (float*)(ws + 34 * MB + 64 * 1024);    // 128 KB
    float*    pooled  = (float*)(ws + 34 * MB + 192 * 1024);   // 8 KB

    ln_mask_kernel<<<BT / 8, 256, 0, stream>>>(embs, labels, ln_g, ln_b,
                                               alpha_p, beta_p, xh, biasArr, mArr);
    cvt_f32_f16_kernel<<<(D3 * DDIM) / 256, 256, 0, stream>>>(in_w, wh, D3 * DDIM);
    qkv_wmma_kernel<<<dim3(BT / 128, D3 / 64), 256, 0, stream>>>(xh, wh, in_b,
                                                                 qh, kh, vh);
    attn_wmma_kernel<<<BDIM * HDIM * (TDIM / 128), 256, 0, stream>>>(
        qh, kh, vh, biasArr, mArr, part);
    pool_reduce_kernel<<<(BDIM * HDIM * DH) / 256, 256, 0, stream>>>(part, pooled);
    outproj_kernel<<<BDIM * 2, 256, 0, stream>>>(pooled, mArr, out_w, out_b,
                                                 (float*)d_out);
}
